// PGAHead_89902255440789
// MI455X (gfx1250) — matrix operand order repaired and checked by reading, not compiled, not run on hardware
//
#include <hip/hip_runtime.h>
#include <stdint.h>

#define N_PTS 4096
#define DIMV 512
#define PROJ_DIM 768

typedef __attribute__((ext_vector_type(16))) __bf16 v16bf;
typedef __attribute__((ext_vector_type(8)))  float  v8f;
typedef __attribute__((ext_vector_type(4)))  unsigned int u32x4;
typedef __attribute__((ext_vector_type(4)))  int v4i;

union Frag32B { u32x4 u[2]; v16bf v; };

#if defined(__gfx1250__) && __has_builtin(__builtin_amdgcn_global_load_async_to_lds_b128)
#define HAVE_ASYNC_LDS 1
typedef __attribute__((address_space(1))) v4i as1_v4i;
typedef __attribute__((address_space(3))) v4i as3_v4i;
#else
#define HAVE_ASYNC_LDS 0
#endif

__device__ __forceinline__ unsigned short f2bf(float f) {
  unsigned int u = __float_as_uint(f);
  return (unsigned short)((u + 0x7FFFu + ((u >> 16) & 1u)) >> 16);
}

// copy 32 bytes (16 bf16) from global to LDS, per thread
__device__ __forceinline__ void tile_copy32B(const unsigned short* __restrict__ g,
                                             unsigned short* __restrict__ l) {
#if HAVE_ASYNC_LDS
  __builtin_amdgcn_global_load_async_to_lds_b128((as1_v4i*)g, (as3_v4i*)l, 0, 0);
  __builtin_amdgcn_global_load_async_to_lds_b128((as1_v4i*)g, (as3_v4i*)l, 16, 0);
#else
  u32x4 a0 = ((const u32x4*)g)[0];
  u32x4 a1 = ((const u32x4*)g)[1];
  ((u32x4*)l)[0] = a0;
  ((u32x4*)l)[1] = a1;
#endif
}

__device__ __forceinline__ void wait_async_all() {
#if HAVE_ASYNC_LDS
#if __has_builtin(__builtin_amdgcn_s_wait_asynccnt)
  __builtin_amdgcn_s_wait_asynccnt(0);
#else
  asm volatile("s_wait_asynccnt 0" ::: "memory");
#endif
#endif
}

__device__ __forceinline__ float blockReduceSum(float v) {
  __shared__ float red[256];
  int t = threadIdx.x;
  red[t] = v; __syncthreads();
  for (int o = 128; o > 0; o >>= 1) {
    if (t < o) red[t] += red[t + o];
    __syncthreads();
  }
  float r = red[0];
  __syncthreads();
  return r;
}

// ---------------- generic fp32 -> bf16 convert ----------------
__global__ __launch_bounds__(256) void k_cvt(const float* __restrict__ x,
                                             unsigned short* __restrict__ y, size_t n) {
  for (size_t p = (size_t)blockIdx.x * blockDim.x + threadIdx.x; p < n;
       p += (size_t)gridDim.x * blockDim.x)
    y[p] = f2bf(x[p]);
}

// ---------------- row L2-normalize + bf16 convert (Gn, G) ----------------
__global__ __launch_bounds__(256) void k_rownorm(const float* __restrict__ X,
                                                 unsigned short* __restrict__ gn,
                                                 unsigned short* __restrict__ g, int dim) {
  int r = blockIdx.x;
  const float* x = X + (size_t)r * dim;
  float s = 0.f;
  for (int c = threadIdx.x; c < dim; c += 256) { float v = x[c]; s += v * v; }
  s = blockReduceSum(s);
  float inv = 1.f / fmaxf(sqrtf(s), 1e-12f);
  for (int c = threadIdx.x; c < dim; c += 256) {
    float v = x[c];
    gn[(size_t)r * dim + c] = f2bf(v * inv);
    g [(size_t)r * dim + c] = f2bf(v);
  }
}

// ---------------- bf16 WMMA NT GEMM: C[MxN] = A[MxK] * B[NxK]^T (+aux) ----------------
// Double-buffered LDS tiles fed by GLOBAL_LOAD_ASYNC_TO_LDS_B128 (ASYNCcnt),
// so the next k-slab DMAs into LDS while the current slab's WMMAs execute.
#define BM 128
#define BN 128
#define BK 32
__global__ __launch_bounds__(256) void k_gemm_nt(const unsigned short* __restrict__ A,
                                                 const unsigned short* __restrict__ B,
                                                 float* __restrict__ C,
                                                 const float* __restrict__ aux,
                                                 int M, int N, int K) {
  __shared__ __align__(16) unsigned short As[2][BM][BK];
  __shared__ __align__(16) unsigned short Bs[2][BM][BK];
  int tid = threadIdx.x;
  int lane = tid & 31, wid = tid >> 5;
  int wm = wid >> 2, wn = wid & 3;                 // 2 x 4 wave grid
  int m0 = blockIdx.y * BM, n0 = blockIdx.x * BN;

  v8f zero = {0.f, 0.f, 0.f, 0.f, 0.f, 0.f, 0.f, 0.f};
  v8f acc[4][2];
  for (int i = 0; i < 4; i++) for (int j = 0; j < 2; j++) acc[i][j] = zero;

  int lrow = tid >> 1;              // 0..127
  int lcol = (tid & 1) * 16;        // 0 or 16

  const unsigned short* ga = A + (size_t)(m0 + lrow) * K + lcol;
  const unsigned short* gb = B + (size_t)(n0 + lrow) * K + lcol;

  // prologue: stage k=0 slab into buffer 0
  tile_copy32B(ga, &As[0][lrow][lcol]);
  tile_copy32B(gb, &Bs[0][lrow][lcol]);
  wait_async_all();
  __syncthreads();

  int lr = lane & 15, lh = lane >> 4;
  for (int k0 = 0; k0 < K; k0 += BK) {
    int cur = (k0 / BK) & 1, nxt = cur ^ 1;
    if (k0 + BK < K) {
      // overlap: DMA next slab while computing current one
      tile_copy32B(ga + k0 + BK, &As[nxt][lrow][lcol]);
      tile_copy32B(gb + k0 + BK, &Bs[nxt][lrow][lcol]);
    }

#pragma unroll
    for (int mt = 0; mt < 4; mt++) {
      Frag32B fa;
      const unsigned short* ap = &As[cur][wm * 64 + mt * 16 + lr][0];
      fa.u[0] = *(const u32x4*)(ap + lh * 8);
      fa.u[1] = *(const u32x4*)(ap + lh * 8 + 16);
#pragma unroll
      for (int nt = 0; nt < 2; nt++) {
        Frag32B fb;
        const unsigned short* bp = &Bs[cur][wn * 32 + nt * 16 + lr][lh * 8];
        fb.u[0] = *(const u32x4*)bp;
        fb.u[1] = *(const u32x4*)(bp + 16);
        acc[mt][nt] = __builtin_amdgcn_wmma_f32_16x16x32_bf16(
            false, fa.v, false, fb.v, (short)0, acc[mt][nt], false, false);
      }
    }

    wait_async_all();
    __syncthreads();
  }

#pragma unroll
  for (int mt = 0; mt < 4; mt++)
#pragma unroll
    for (int nt = 0; nt < 2; nt++) {
      int col  = n0 + wn * 32 + nt * 16 + lr;
      int row0 = m0 + wm * 64 + mt * 16 + lh * 8;
#pragma unroll
      for (int v = 0; v < 8; v++) {
        size_t off = (size_t)(row0 + v) * N + col;
        float val = acc[mt][nt][v];
        if (aux) val += aux[off];
        C[off] = val;
      }
    }
}

// ---------------- fp32 -> bf16 transpose (for S @ X operands) ----------------
__global__ __launch_bounds__(256) void k_cvt_transpose(const float* __restrict__ X,
                                                       unsigned short* __restrict__ Xt,
                                                       int rows, int cols) {
  __shared__ float tile[32][33];
  int bx = blockIdx.x * 32, by = blockIdx.y * 32;
  int tx = threadIdx.x & 31, ty = threadIdx.x >> 5;  // 32 x 8
  for (int i = ty; i < 32; i += 8)
    tile[i][tx] = X[(size_t)(by + i) * cols + bx + tx];
  __syncthreads();
  for (int i = ty; i < 32; i += 8)
    Xt[(size_t)(bx + i) * rows + by + tx] = f2bf(tile[tx][i]);
}

// ---------------- same-label top-8 per row ----------------
__global__ __launch_bounds__(256) void k_topk8(const float* __restrict__ sim,
                                               const int* __restrict__ labels,
                                               int* __restrict__ idx, int n) {
  int i = blockIdx.x;
  __shared__ float sv[256];
  __shared__ int   si[256];
  __shared__ int   chosen[8];
  int li = labels[i];
  const float* row = sim + (size_t)i * n;
  for (int p = 0; p < 8; p++) {
    float best = -3e38f; int bj = -1;
    for (int j = threadIdx.x; j < n; j += 256) {
      if (j == i || labels[j] != li) continue;
      bool used = false;
      for (int q = 0; q < p; q++) if (chosen[q] == j) used = true;
      if (used) continue;
      float v = row[j];
      if (v > best) { best = v; bj = j; }
    }
    sv[threadIdx.x] = best; si[threadIdx.x] = bj;
    __syncthreads();
    for (int s = 128; s > 0; s >>= 1) {
      if (threadIdx.x < s && sv[threadIdx.x + s] > sv[threadIdx.x]) {
        sv[threadIdx.x] = sv[threadIdx.x + s];
        si[threadIdx.x] = si[threadIdx.x + s];
      }
      __syncthreads();
    }
    if (threadIdx.x == 0) chosen[p] = si[0];
    __syncthreads();
  }
  if (threadIdx.x < 8) idx[i * 8 + threadIdx.x] = chosen[threadIdx.x];
}

// ---------------- scatter symmetric KNN graph (Sg pre-zeroed) ----------------
__global__ void k_build_graph(const float* __restrict__ sim, const int* __restrict__ idx,
                              float* __restrict__ Sg, float alpha, int n) {
  int i = blockIdx.x, t = threadIdx.x;  // block of 32
  if (t < 8) {
    int j = idx[i * 8 + t];
    if (j >= 0) {
      float v = sim[(size_t)i * n + j];
      v = fminf(fmaxf(v, -1.f + 1e-8f), 1.f - 1e-8f);
      v = fmaxf(v, 0.f) * alpha;
      Sg[(size_t)i * n + j] = v;
      Sg[(size_t)j * n + i] = v;   // symmetric union; duplicate writes store same value
    }
  } else if (t == 8) {
    Sg[(size_t)i * n + i] = 1e-6f;
  }
}

__global__ __launch_bounds__(256) void k_rowsum(const float* __restrict__ Sg,
                                                float* __restrict__ dinv, int n) {
  int r = blockIdx.x; float s = 0.f;
  for (int c = threadIdx.x; c < n; c += 256) s += Sg[(size_t)r * n + c];
  s = blockReduceSum(s);
  if (threadIdx.x == 0) dinv[r] = rsqrtf(fmaxf(s, 1e-8f));
}

__global__ __launch_bounds__(256) void k_sym_norm(const float* __restrict__ Sg,
                                                  const float* __restrict__ dinv,
                                                  float* __restrict__ S,
                                                  unsigned short* __restrict__ Sb, int n) {
  int row = blockIdx.y;
  int col = blockIdx.x * 256 + threadIdx.x;
  size_t p = (size_t)row * n + col;
  float v = dinv[row] * Sg[p] * dinv[col];
  S[p] = v;
  Sb[p] = f2bf(v);
}

// ---------------- BatchNorm (training stats) + ReLU + bf16 ----------------
__global__ __launch_bounds__(256) void k_bn_stats(const float* __restrict__ Z,
                                                  float* __restrict__ mean,
                                                  float* __restrict__ var, int rows, int cols) {
  int c = blockIdx.x;
  float s = 0.f, s2 = 0.f;
  for (int r = threadIdx.x; r < rows; r += 256) {
    float v = Z[(size_t)r * cols + c];
    s += v; s2 += v * v;
  }
  s  = blockReduceSum(s);
  s2 = blockReduceSum(s2);
  if (threadIdx.x == 0) {
    float m = s / rows;
    mean[c] = m;
    var[c] = s2 / rows - m * m;   // biased, torch-BN style
  }
}

__global__ __launch_bounds__(256) void k_bn_relu_cvt(const float* __restrict__ Z,
                                                     unsigned short* __restrict__ out,
                                                     const float* __restrict__ mean,
                                                     const float* __restrict__ var,
                                                     const float* __restrict__ gamma,
                                                     const float* __restrict__ beta, int cols) {
  int row = blockIdx.y, col = blockIdx.x * 256 + threadIdx.x;
  size_t p = (size_t)row * cols + col;
  float x = Z[p];
  float y = (x - mean[col]) * rsqrtf(var[col] + 1e-5f) * gamma[col] + beta[col];
  out[p] = f2bf(fmaxf(y, 0.f));
}

__global__ __launch_bounds__(256) void k_rowl2norm(float* __restrict__ P, int cols) {
  int r = blockIdx.x;
  float* row = P + (size_t)r * cols;
  float s = 0.f;
  for (int c = threadIdx.x; c < cols; c += 256) { float v = row[c]; s += v * v; }
  s = blockReduceSum(s);
  float inv = 1.f / fmaxf(sqrtf(s), 1e-12f);
  for (int c = threadIdx.x; c < cols; c += 256) row[c] *= inv;
}

// ---------------- loss reductions ----------------
__global__ __launch_bounds__(256) void k_sqdiff(const float* __restrict__ A,
                                                const float* __restrict__ B,
                                                float* __restrict__ acc, size_t n) {
  float s = 0.f;
  for (size_t p = (size_t)blockIdx.x * blockDim.x + threadIdx.x; p < n;
       p += (size_t)gridDim.x * blockDim.x) {
    float d = A[p] - B[p]; s += d * d;
  }
  s = blockReduceSum(s);
  if (threadIdx.x == 0) atomicAdd(acc, s);
}

__global__ __launch_bounds__(256) void k_idea(const float* __restrict__ S,
                                              const int* __restrict__ labels,
                                              float* __restrict__ acc, int n) {
  int row = blockIdx.y, col = blockIdx.x * 256 + threadIdx.x;
  float tgt = (labels[row] == labels[col]) ? 0.99f : 0.f;
  float d = S[(size_t)row * n + col] - tgt;
  float s = blockReduceSum(d * d);
  if (threadIdx.x == 0) atomicAdd(acc, s);
}

__global__ void k_finalize(const float* __restrict__ acc, float* __restrict__ out) {
  float nn = 4096.f * 4096.f;
  float lossK = (acc[0] + acc[1]) / nn;
  float lossZ = (acc[2] + acc[3]) / (4096.f * 768.f);
  float lossI = acc[4] / nn;
  out[0] = 64.f * lossK + 16.f * lossZ + lossI;
}

// ---------------- orchestration ----------------
extern "C" void kernel_launch(void* const* d_in, const int* in_sizes, int n_in,
                              void* d_out, int out_size, void* d_ws, size_t ws_size,
                              hipStream_t stream) {
  (void)in_sizes; (void)n_in; (void)out_size; (void)ws_size;
  const float* feats  = (const float*)d_in[0];
  const int*   labels = (const int*)d_in[1];
  const float* fc1    = (const float*)d_in[2];
  const float* fc2    = (const float*)d_in[3];
  const float* bng    = (const float*)d_in[4];
  const float* bnb    = (const float*)d_in[5];
  const float* proj   = (const float*)d_in[6];
  float* out = (float*)d_out;

  char* w = (char*)d_ws; size_t off = 0;
  auto alloc = [&](size_t bytes) -> char* {
    char* p = w + off; off += (bytes + 255) & ~(size_t)255; return p;
  };
  const size_t NN = (size_t)N_PTS * N_PTS;

  float* Sarr[3];
  for (int l = 0; l < 3; l++) Sarr[l] = (float*)alloc(NN * 4);
  float* sim = (float*)alloc(NN * 4);
  float* Sg  = (float*)alloc(NN * 4);
  unsigned short* Sb  = (unsigned short*)alloc(NN * 2);
  unsigned short* Gn  = (unsigned short*)alloc((size_t)N_PTS * DIMV * 2);
  unsigned short* Gb  = (unsigned short*)alloc((size_t)N_PTS * DIMV * 2);
  unsigned short* W1b = (unsigned short*)alloc((size_t)DIMV * DIMV * 2);
  unsigned short* W2b = (unsigned short*)alloc((size_t)DIMV * DIMV * 2);
  unsigned short* Pjb = (unsigned short*)alloc((size_t)PROJ_DIM * DIMV * 2);
  float* H   = (float*)alloc((size_t)N_PTS * DIMV * 4);
  unsigned short* Ht  = (unsigned short*)alloc((size_t)DIMV * N_PTS * 2);
  float* Zp  = (float*)alloc((size_t)N_PTS * DIMV * 4);
  unsigned short* Z1b = (unsigned short*)alloc((size_t)N_PTS * DIMV * 2);
  unsigned short* Zb  = (unsigned short*)alloc((size_t)N_PTS * DIMV * 2);
  float* Parr[3];
  for (int l = 0; l < 3; l++) Parr[l] = (float*)alloc((size_t)N_PTS * PROJ_DIM * 4);
  float* dinv = (float*)alloc(N_PTS * 4);
  int*   idx  = (int*)alloc(N_PTS * 8 * 4);
  float* mean = (float*)alloc(DIMV * 4);
  float* var  = (float*)alloc(DIMV * 4);
  float* accs = (float*)alloc(8 * 4);

  (void)hipMemsetAsync(accs, 0, 8 * 4, stream);
  k_cvt<<<384, 256, 0, stream>>>(proj, Pjb, (size_t)PROJ_DIM * DIMV);

  const float ALPHA[3] = {1.0f, 1.1f, 1.2f};   // linspace(1.0, 1.2, 3); BETA == 0 -> inter branch dead
  for (int l = 0; l < 3; l++) {
    const float* G = feats + (size_t)l * N_PTS * DIMV;
    k_rownorm<<<N_PTS, 256, 0, stream>>>(G, Gn, Gb, DIMV);
    // sim = Gn * Gn^T   (WMMA, K=512)
    k_gemm_nt<<<dim3(N_PTS / BN, N_PTS / BM), 256, 0, stream>>>(Gn, Gn, sim, nullptr, N_PTS, N_PTS, DIMV);
    k_topk8<<<N_PTS, 256, 0, stream>>>(sim, labels, idx, N_PTS);
    (void)hipMemsetAsync(Sg, 0, NN * 4, stream);
    k_build_graph<<<N_PTS, 32, 0, stream>>>(sim, idx, Sg, ALPHA[l], N_PTS);
    k_rowsum<<<N_PTS, 256, 0, stream>>>(Sg, dinv, N_PTS);
    k_sym_norm<<<dim3(N_PTS / 256, N_PTS), 256, 0, stream>>>(Sg, dinv, Sarr[l], Sb, N_PTS);
    // layer weights -> bf16
    k_cvt<<<256, 256, 0, stream>>>(fc1 + (size_t)l * DIMV * DIMV, W1b, (size_t)DIMV * DIMV);
    k_cvt<<<256, 256, 0, stream>>>(fc2 + (size_t)l * DIMV * DIMV, W2b, (size_t)DIMV * DIMV);
    // H1 = G * W1^T
    k_gemm_nt<<<dim3(DIMV / BN, N_PTS / BM), 256, 0, stream>>>(Gb, W1b, H, nullptr, N_PTS, DIMV, DIMV);
    k_cvt_transpose<<<dim3(DIMV / 32, N_PTS / 32), 256, 0, stream>>>(H, Ht, N_PTS, DIMV);
    // Zpre = S * H1   (WMMA, K=4096)
    k_gemm_nt<<<dim3(DIMV / BN, N_PTS / BM), 256, 0, stream>>>(Sb, Ht, Zp, nullptr, N_PTS, DIMV, N_PTS);
    k_bn_stats<<<DIMV, 256, 0, stream>>>(Zp, mean, var, N_PTS, DIMV);
    k_bn_relu_cvt<<<dim3(DIMV / 256, N_PTS), 256, 0, stream>>>(Zp, Z1b, mean, var,
                                                               bng + l * DIMV, bnb + l * DIMV, DIMV);
    // H2 = Z1 * W2^T
    k_gemm_nt<<<dim3(DIMV / BN, N_PTS / BM), 256, 0, stream>>>(Z1b, W2b, H, nullptr, N_PTS, DIMV, DIMV);
    k_cvt_transpose<<<dim3(DIMV / 32, N_PTS / 32), 256, 0, stream>>>(H, Ht, N_PTS, DIMV);
    // Z = S * H2 + G   (WMMA + fused residual epilogue)
    k_gemm_nt<<<dim3(DIMV / BN, N_PTS / BM), 256, 0, stream>>>(Sb, Ht, Zp, G, N_PTS, DIMV, N_PTS);
    k_cvt<<<2048, 256, 0, stream>>>(Zp, Zb, (size_t)N_PTS * DIMV);
    // P = Z * proj^T, then row-L2-normalize
    k_gemm_nt<<<dim3(PROJ_DIM / BN, N_PTS / BM), 256, 0, stream>>>(Zb, Pjb, Parr[l], nullptr,
                                                                   N_PTS, PROJ_DIM, DIMV);
    k_rowl2norm<<<N_PTS, 256, 0, stream>>>(Parr[l], PROJ_DIM);
  }

  // losses: K = S (t_diff == 1), stop_gradient irrelevant for the forward value
  k_sqdiff<<<2048, 256, 0, stream>>>(Sarr[0], Sarr[1], accs + 0, NN);
  k_sqdiff<<<2048, 256, 0, stream>>>(Sarr[1], Sarr[2], accs + 1, NN);
  k_sqdiff<<<1024, 256, 0, stream>>>(Parr[0], Parr[1], accs + 2, (size_t)N_PTS * PROJ_DIM);
  k_sqdiff<<<1024, 256, 0, stream>>>(Parr[1], Parr[2], accs + 3, (size_t)N_PTS * PROJ_DIM);
  k_idea<<<dim3(N_PTS / 256, N_PTS), 256, 0, stream>>>(Sarr[2], labels, accs + 4, N_PTS);
  k_finalize<<<1, 1, 0, stream>>>(accs, out);
}
